// SelfAttention_30434138259803
// MI455X (gfx1250) — matrix operand order repaired and checked by reading, not compiled, not run on hardware
//
#include <hip/hip_runtime.h>
#include <hip/hip_bf16.h>
#include <hip/hip_fp16.h>

typedef __attribute__((ext_vector_type(16))) _Float16 v16h;
typedef __attribute__((ext_vector_type(8)))  float    v8f;
typedef __attribute__((ext_vector_type(4)))  unsigned int tdm_v4u;
typedef __attribute__((ext_vector_type(8)))  int          tdm_v8i;
typedef __attribute__((ext_vector_type(4)))  int          tdm_v4i;

#define NHEAD  16
#define DHEAD  64
#define HID    1024
#define SEQ    2048
#define BATCH  4
#define NROWS  (BATCH * SEQ)   // 8192

// ---------------------------------------------------------------------------
// WMMA fragment loaders (CDNA5 ISA §7.12.2, wave32)
// A 16x32 f16: lane lo = row M; j0..7 -> K = j + 8*hi ; j8..15 -> K = j+8+8*hi
// B 32x16 f16: lane lo = col N; j0..7 -> K = j + 8*hi ; j8..15 -> K = j+8+8*hi
// C/D 16x16 f32: VGPR r, lane -> (M = r + 8*hi, N = lo)
// ---------------------------------------------------------------------------
__device__ __forceinline__ v16h frag_a(const _Float16* base, int stride) {
  const int lane = threadIdx.x & 31;
  const int lo = lane & 15, hi = lane >> 4;
  const _Float16* p = base + lo * stride;
  union { v16h v; uint4 u[2]; } r;
  r.u[0] = *(const uint4*)(p + 8 * hi);
  r.u[1] = *(const uint4*)(p + 16 + 8 * hi);
  return r.v;
}

__device__ __forceinline__ v16h frag_b(const _Float16* base, int stride) {
  const int lane = threadIdx.x & 31;
  const int lo = lane & 15, hi = lane >> 4;
  const _Float16* p = base + lo * stride + 8 * hi;
  union { v16h v; uint4 u[2]; } r;
  r.u[0] = *(const uint4*)(p);
  r.u[1] = *(const uint4*)(p + 16);
  return r.v;
}

__device__ __forceinline__ v8f wmma_f16(v16h a, v16h b, v8f c) {
  return __builtin_amdgcn_wmma_f32_16x16x32_f16(false, a, false, b, (short)0, c,
                                                false, false);
}

// ---------------------------------------------------------------------------
// TDM: issue a 2D tile load (64 rows x 64 halves = 64 x 128B) into LDS.
// D# per CDNA5 ISA:
//   group0: count=1 | lds_addr | global_addr | type=2
//   group1: data_size=8B, tensor_dim0=K/4 units, tensor_dim1=64,
//           tile_dim0=16 units (128B row), tile_dim1=64, stride0=K/4 units
// Tracked on TENSORcnt; issued by one wave only (EXEC is ignored by TDM).
// This toolchain exposes the 6-arg builtin:
//   (uint32x4 g0, int32x8 g1, int32x4 g2, int32x4 g3, int32x8 g4, i32 cpol)
// ---------------------------------------------------------------------------
#if __has_builtin(__builtin_amdgcn_tensor_load_to_lds)
#define HAVE_TDM 1
__device__ __forceinline__ void tdm_issue_tile64x64(unsigned int lds_byte,
                                                    const _Float16* g,
                                                    int Khalves) {
  const unsigned long long ga = (unsigned long long)(uintptr_t)g;
  const unsigned int u0 = (unsigned int)(Khalves >> 2);  // row len, 8B units
  tdm_v4u g0;
  g0.x = 1u;                                             // count = 1
  g0.y = lds_byte;                                       // lds_addr
  g0.z = (unsigned int)ga;                               // global_addr lo
  g0.w = (unsigned int)((ga >> 32) & 0x01FFFFFFull) | (2u << 30);  // hi | type=2
  tdm_v8i g1;
  g1[0] = (int)(3u << 16);                       // wg_mask=0, data_size=8B
  g1[1] = (int)((u0 & 0xFFFFu) << 16);           // tensor_dim0[15:0]
  g1[2] = (int)((u0 >> 16) | (64u << 16));       // tensor_dim0[31:16] | dim1 lo
  g1[3] = (int)(16u << 16);                      // dim1 hi=0 | tile_dim0=16
  g1[4] = (int)64u;                              // tile_dim1=64 | tile_dim2=0
  g1[5] = (int)u0;                               // stride0[31:0] (8B units)
  g1[6] = 0;                                     // stride0[47:32]=0 | stride1 lo
  g1[7] = 0;
  tdm_v4i z4 = {};
  tdm_v8i z8 = {};
  __builtin_amdgcn_tensor_load_to_lds(g0, g1, z4, z4, z8, 0);
}
#else
#define HAVE_TDM 0
#endif

// ---------------------------------------------------------------------------
// fp32 -> f16 elementwise convert
// ---------------------------------------------------------------------------
__global__ __launch_bounds__(256)
void cvt_f32_to_f16(const float* __restrict__ in, _Float16* __restrict__ out, int n4) {
  int i = blockIdx.x * 256 + threadIdx.x;
  if (i >= n4) return;
  float4 v = *(const float4*)(in + (size_t)i * 4);
  union { uint2 u; _Float16 h[4]; } o;
  o.h[0] = (_Float16)v.x; o.h[1] = (_Float16)v.y;
  o.h[2] = (_Float16)v.z; o.h[3] = (_Float16)v.w;
  *(uint2*)(out + (size_t)i * 4) = o.u;
}

// ---------------------------------------------------------------------------
// C(f16, MxN) = A(f16, MxK) @ W(f16, NxK)^T + bias(f32)
// block tile 64x64, 256 threads = 8 waves as 4(M) x 2(N); k-step 64.
// TDM double-buffered LDS pipeline when available.
// ---------------------------------------------------------------------------
__global__ __launch_bounds__(256)
void gemm_f16(const _Float16* __restrict__ A, const _Float16* __restrict__ W,
              const float* __restrict__ bias, _Float16* __restrict__ C,
              int M, int N, int K) {
  __shared__ __align__(16) _Float16 sA[2][64 * 64];
  __shared__ __align__(16) _Float16 sB[2][64 * 64];

  const int tid = threadIdx.x;
  const int wave = tid >> 5;
  const int wm = wave >> 1;          // 0..3
  const int wn = wave & 1;           // 0..1
  const int m0 = blockIdx.y * 64;
  const int n0 = blockIdx.x * 64;

  v8f acc0 = {}, acc1 = {};
  const int nk = K >> 6;

#if HAVE_TDM
  const unsigned int ldsA[2] = {(unsigned int)(uintptr_t)&sA[0][0],
                                (unsigned int)(uintptr_t)&sA[1][0]};
  const unsigned int ldsB[2] = {(unsigned int)(uintptr_t)&sB[0][0],
                                (unsigned int)(uintptr_t)&sB[1][0]};
  if (wave == 0) {
    tdm_issue_tile64x64(ldsA[0], A + (size_t)m0 * K, K);
    tdm_issue_tile64x64(ldsB[0], W + (size_t)n0 * K, K);
  }
  for (int kt = 0; kt < nk; ++kt) {
    const int buf = kt & 1;
    if (wave == 0) {
      if (kt + 1 < nk) {
        const int k0n = (kt + 1) << 6;
        tdm_issue_tile64x64(ldsA[buf ^ 1], A + (size_t)m0 * K + k0n, K);
        tdm_issue_tile64x64(ldsB[buf ^ 1], W + (size_t)n0 * K + k0n, K);
        __builtin_amdgcn_s_wait_tensorcnt(2);   // current buf's 2 DMAs done
      } else {
        __builtin_amdgcn_s_wait_tensorcnt(0);
      }
    }
    __syncthreads();
#pragma unroll
    for (int kk = 0; kk < 2; ++kk) {
      v16h a  = frag_a(&sA[buf][wm * 16 * 64 + kk * 32], 64);
      acc0 = wmma_f16(a, frag_b(&sB[buf][(wn * 32 + 0)  * 64 + kk * 32], 64), acc0);
      acc1 = wmma_f16(a, frag_b(&sB[buf][(wn * 32 + 16) * 64 + kk * 32], 64), acc1);
    }
    __syncthreads();
  }
#else
  const int lrow = tid >> 2;          // 0..63
  const int lcol = (tid & 3) * 16;    // 0,16,32,48
  for (int kt = 0; kt < nk; ++kt) {
    const int k0 = kt << 6;
    {
      const _Float16* g = A + (size_t)(m0 + lrow) * K + k0 + lcol;
      *(uint4*)&sA[0][lrow * 64 + lcol]     = *(const uint4*)g;
      *(uint4*)&sA[0][lrow * 64 + lcol + 8] = *(const uint4*)(g + 8);
      if (k0 + 64 < K) __builtin_prefetch(g + 64, 0, 0);
    }
    {
      const _Float16* g = W + (size_t)(n0 + lrow) * K + k0 + lcol;
      *(uint4*)&sB[0][lrow * 64 + lcol]     = *(const uint4*)g;
      *(uint4*)&sB[0][lrow * 64 + lcol + 8] = *(const uint4*)(g + 8);
    }
    __syncthreads();
#pragma unroll
    for (int kk = 0; kk < 2; ++kk) {
      v16h a  = frag_a(&sA[0][wm * 16 * 64 + kk * 32], 64);
      acc0 = wmma_f16(a, frag_b(&sB[0][(wn * 32 + 0)  * 64 + kk * 32], 64), acc0);
      acc1 = wmma_f16(a, frag_b(&sB[0][(wn * 32 + 16) * 64 + kk * 32], 64), acc1);
    }
    __syncthreads();
  }
#endif

  const int lane = tid & 31, lo = lane & 15, hi = lane >> 4;
  const int gm = m0 + wm * 16 + 8 * hi;
  {
    int gn = n0 + wn * 32 + lo;
    float bv = bias[gn];
    _Float16* cp = C + (size_t)gm * N + gn;
#pragma unroll
    for (int r = 0; r < 8; ++r) cp[(size_t)r * N] = (_Float16)(acc0[r] + bv);
    gn += 16;
    bv = bias[gn];
    cp = C + (size_t)gm * N + gn;
#pragma unroll
    for (int r = 0; r < 8; ++r) cp[(size_t)r * N] = (_Float16)(acc1[r] + bv);
  }
}

// ---------------------------------------------------------------------------
// QK RMS-norm + head split. qkv: (NROWS, 3*HID) f16.
// qh/kh/vh: (B, NHEAD, SEQ, DHEAD) f16. q gets g*sqrt(d)/||q|| * d^-1/2.
// ---------------------------------------------------------------------------
__global__ __launch_bounds__(256)
void qk_norm(const _Float16* __restrict__ qkv, const float* __restrict__ g_q,
             const float* __restrict__ g_k, _Float16* __restrict__ qh,
             _Float16* __restrict__ kh, _Float16* __restrict__ vh) {
  __shared__ float red[2][8];
  const int row = blockIdx.x;              // b*SEQ + l
  const int b = row >> 11, l = row & (SEQ - 1);
  const _Float16* p = qkv + (size_t)row * (3 * HID);
  const int t = threadIdx.x;

  float qv[4], kv[4], vv[4];
  float sq = 0.f, sk = 0.f;
  {
    union { uint2 u; _Float16 h[4]; } dq, dk, dv;
    const int c = t * 4;
    dq.u = *(const uint2*)(p + c);
    dk.u = *(const uint2*)(p + HID + c);
    dv.u = *(const uint2*)(p + 2 * HID + c);
#pragma unroll
    for (int j = 0; j < 4; ++j) {
      qv[j] = (float)dq.h[j]; kv[j] = (float)dk.h[j]; vv[j] = (float)dv.h[j];
      sq += qv[j] * qv[j];    sk += kv[j] * kv[j];
    }
  }
#pragma unroll
  for (int m = 16; m; m >>= 1) { sq += __shfl_xor(sq, m); sk += __shfl_xor(sk, m); }
  if ((t & 31) == 0) { red[0][t >> 5] = sq; red[1][t >> 5] = sk; }
  __syncthreads();
  float tq = 0.f, tk = 0.f;
#pragma unroll
  for (int i = 0; i < 8; ++i) { tq += red[0][i]; tk += red[1][i]; }

  const float sq_scale = 32.0f / fmaxf(sqrtf(tq), 1e-12f) * 0.125f;  // *d^-0.5
  const float sk_scale = 32.0f / fmaxf(sqrtf(tk), 1e-12f);

  const int c0 = t * 4;
  const int h = c0 >> 6, d0 = c0 & 63;
  const size_t base = ((size_t)(b * NHEAD + h) * SEQ + l) * DHEAD + d0;
#pragma unroll
  for (int j = 0; j < 4; ++j) {
    const int c = c0 + j;
    qh[base + j] = (_Float16)(qv[j] * g_q[c] * sq_scale);
    kh[base + j] = (_Float16)(kv[j] * g_k[c] * sk_scale);
    vh[base + j] = (_Float16)vv[j];
  }
}

// ---------------------------------------------------------------------------
// Flash attention. Block = (b,h) x 128 query rows; 8 waves x 16 rows each.
// out: (B, SEQ, HID) f16.
// ---------------------------------------------------------------------------
__global__ __launch_bounds__(256)
void attn_kernel(const _Float16* __restrict__ qh, const _Float16* __restrict__ kh,
                 const _Float16* __restrict__ vh, _Float16* __restrict__ out) {
  __shared__ __align__(16) _Float16 sK[64 * 64];      // [key][d]
  __shared__ __align__(16) _Float16 sVt[64 * 64];     // [d][key]
  __shared__ __align__(16) _Float16 sP[8 * 16 * 64];  // per-wave P staging

  const int tid = threadIdx.x;
  const int wave = tid >> 5;
  const int lane = tid & 31, lo = lane & 15, hi = lane >> 4;

  const int qt = blockIdx.x;          // 0..15
  const int bh = blockIdx.y;          // 0..63 = b*NHEAD + h
  const int b = bh >> 4, h = bh & 15;

  const _Float16* Q  = qh + (size_t)bh * SEQ * DHEAD;
  const _Float16* Kg = kh + (size_t)bh * SEQ * DHEAD;
  const _Float16* Vg = vh + (size_t)bh * SEQ * DHEAD;

  const int m0 = qt * 128 + wave * 16;

  const v16h qa0 = frag_a(Q + (size_t)m0 * DHEAD + 0, DHEAD);
  const v16h qa1 = frag_a(Q + (size_t)m0 * DHEAD + 32, DHEAD);

  float mi[8], li[8];
#pragma unroll
  for (int r = 0; r < 8; ++r) { mi[r] = -1e30f; li[r] = 0.f; }
  v8f o0 = {}, o1 = {}, o2 = {}, o3 = {};

  const int lrow = tid >> 2;          // 0..63
  const int lcol = (tid & 3) * 16;    // 0,16,32,48

  for (int kt = 0; kt < SEQ / 64; ++kt) {
    {
      const _Float16* g = Kg + (size_t)(kt * 64 + lrow) * DHEAD + lcol;
      *(uint4*)&sK[lrow * 64 + lcol]     = *(const uint4*)g;
      *(uint4*)&sK[lrow * 64 + lcol + 8] = *(const uint4*)(g + 8);
    }
    {
      const _Float16* g = Vg + (size_t)(kt * 64 + lrow) * DHEAD + lcol;
      union { uint4 u; _Float16 h[8]; } d0, d1;
      d0.u = *(const uint4*)g;
      d1.u = *(const uint4*)(g + 8);
#pragma unroll
      for (int j = 0; j < 8; ++j) sVt[(lcol + j) * 64 + lrow]     = d0.h[j];
#pragma unroll
      for (int j = 0; j < 8; ++j) sVt[(lcol + 8 + j) * 64 + lrow] = d1.h[j];
    }
    __syncthreads();

    v8f s0 = {}, s1 = {}, s2 = {}, s3 = {};
    s0 = wmma_f16(qa0, frag_b(&sK[0 * 16 * 64 + 0],  64), s0);
    s0 = wmma_f16(qa1, frag_b(&sK[0 * 16 * 64 + 32], 64), s0);
    s1 = wmma_f16(qa0, frag_b(&sK[1 * 16 * 64 + 0],  64), s1);
    s1 = wmma_f16(qa1, frag_b(&sK[1 * 16 * 64 + 32], 64), s1);
    s2 = wmma_f16(qa0, frag_b(&sK[2 * 16 * 64 + 0],  64), s2);
    s2 = wmma_f16(qa1, frag_b(&sK[2 * 16 * 64 + 32], 64), s2);
    s3 = wmma_f16(qa0, frag_b(&sK[3 * 16 * 64 + 0],  64), s3);
    s3 = wmma_f16(qa1, frag_b(&sK[3 * 16 * 64 + 32], 64), s3);

    float alpha[8];
#pragma unroll
    for (int r = 0; r < 8; ++r) {
      float rm = fmaxf(fmaxf(s0[r], s1[r]), fmaxf(s2[r], s3[r]));
#pragma unroll
      for (int m = 8; m; m >>= 1) rm = fmaxf(rm, __shfl_xor(rm, m));
      const float mn = fmaxf(mi[r], rm);
      alpha[r] = __expf(mi[r] - mn);
      s0[r] = __expf(s0[r] - mn); s1[r] = __expf(s1[r] - mn);
      s2[r] = __expf(s2[r] - mn); s3[r] = __expf(s3[r] - mn);
      float rs = s0[r] + s1[r] + s2[r] + s3[r];
#pragma unroll
      for (int m = 8; m; m >>= 1) rs += __shfl_xor(rs, m);
      li[r] = li[r] * alpha[r] + rs;
      mi[r] = mn;
    }
#pragma unroll
    for (int r = 0; r < 8; ++r) {
      o0[r] *= alpha[r]; o1[r] *= alpha[r]; o2[r] *= alpha[r]; o3[r] *= alpha[r];
    }

    _Float16* pw = &sP[wave * 16 * 64];
#pragma unroll
    for (int r = 0; r < 8; ++r) {
      const int prow = r + 8 * hi;
      pw[prow * 64 + 0  + lo] = (_Float16)s0[r];
      pw[prow * 64 + 16 + lo] = (_Float16)s1[r];
      pw[prow * 64 + 32 + lo] = (_Float16)s2[r];
      pw[prow * 64 + 48 + lo] = (_Float16)s3[r];
    }
    asm volatile("s_wait_dscnt 0" ::: "memory");  // DS store -> DS frag load

    const v16h p0 = frag_a(pw + 0, 64);
    const v16h p1 = frag_a(pw + 32, 64);
    o0 = wmma_f16(p0, frag_b(&sVt[0 * 16 * 64 + 0],  64), o0);
    o0 = wmma_f16(p1, frag_b(&sVt[0 * 16 * 64 + 32], 64), o0);
    o1 = wmma_f16(p0, frag_b(&sVt[1 * 16 * 64 + 0],  64), o1);
    o1 = wmma_f16(p1, frag_b(&sVt[1 * 16 * 64 + 32], 64), o1);
    o2 = wmma_f16(p0, frag_b(&sVt[2 * 16 * 64 + 0],  64), o2);
    o2 = wmma_f16(p1, frag_b(&sVt[2 * 16 * 64 + 32], 64), o2);
    o3 = wmma_f16(p0, frag_b(&sVt[3 * 16 * 64 + 0],  64), o3);
    o3 = wmma_f16(p1, frag_b(&sVt[3 * 16 * 64 + 32], 64), o3);
    __syncthreads();
  }

#pragma unroll
  for (int r = 0; r < 8; ++r) {
    const float inv = 1.0f / li[r];
    const int l = m0 + r + 8 * hi;
    _Float16* op = out + ((size_t)(b * SEQ + l)) * HID + h * DHEAD;
    op[0  + lo] = (_Float16)(o0[r] * inv);
    op[16 + lo] = (_Float16)(o1[r] * inv);
    op[32 + lo] = (_Float16)(o2[r] * inv);
    op[48 + lo] = (_Float16)(o3[r] * inv);
  }
}

// ---------------------------------------------------------------------------
// final RMS-norm: out_f32 = in / max(||in||,eps) * g * sqrt(1024)
// ---------------------------------------------------------------------------
__global__ __launch_bounds__(256)
void out_norm(const _Float16* __restrict__ in, const float* __restrict__ g,
              float* __restrict__ out) {
  __shared__ float red[8];
  const int row = blockIdx.x;
  const _Float16* p = in + (size_t)row * HID;
  const int t = threadIdx.x;
  float v[4];
  float ss = 0.f;
  {
    union { uint2 u; _Float16 h[4]; } d;
    d.u = *(const uint2*)(p + t * 4);
#pragma unroll
    for (int j = 0; j < 4; ++j) { v[j] = (float)d.h[j]; ss += v[j] * v[j]; }
  }
#pragma unroll
  for (int m = 16; m; m >>= 1) ss += __shfl_xor(ss, m);
  if ((t & 31) == 0) red[t >> 5] = ss;
  __syncthreads();
  float tot = 0.f;
#pragma unroll
  for (int i = 0; i < 8; ++i) tot += red[i];
  const float s = 32.0f / fmaxf(sqrtf(tot), 1e-12f);
#pragma unroll
  for (int j = 0; j < 4; ++j) {
    const int c = t * 4 + j;
    out[(size_t)row * HID + c] = v[j] * s * g[c];
  }
}

// ---------------------------------------------------------------------------
extern "C" void kernel_launch(void* const* d_in, const int* in_sizes, int n_in,
                              void* d_out, int out_size, void* d_ws, size_t ws_size,
                              hipStream_t stream) {
  const float* x     = (const float*)d_in[0];
  const float* Wqkv  = (const float*)d_in[1];
  const float* bqkv  = (const float*)d_in[2];
  const float* Wout  = (const float*)d_in[3];
  const float* bout  = (const float*)d_in[4];
  const float* g_q   = (const float*)d_in[5];
  const float* g_k   = (const float*)d_in[6];
  const float* g_out = (const float*)d_in[7];

  char* ws = (char*)d_ws;
  const size_t MB = 1024 * 1024;
  _Float16* x16    = (_Float16*)(ws + 0);          // 16 MB
  _Float16* wqkv16 = (_Float16*)(ws + 16 * MB);    //  6 MB
  _Float16* wout16 = (_Float16*)(ws + 22 * MB);    //  2 MB
  _Float16* qkv16  = (_Float16*)(ws + 24 * MB);    // 48 MB
  _Float16* qh     = (_Float16*)(ws + 72 * MB);    // 16 MB
  _Float16* kh     = (_Float16*)(ws + 88 * MB);    // 16 MB
  _Float16* vh     = (_Float16*)(ws + 104 * MB);   // 16 MB
  _Float16* attn16 = (_Float16*)(ws + 0);          // reuse x16 region
  _Float16* proj16 = (_Float16*)(ws + 24 * MB);    // reuse qkv16 region

  // 1) fp32 -> f16 conversions (x, Wqkv, Wout)
  {
    int n4 = NROWS * HID / 4;
    cvt_f32_to_f16<<<(n4 + 255) / 256, 256, 0, stream>>>(x, x16, n4);
    n4 = 3 * HID * HID / 4;
    cvt_f32_to_f16<<<(n4 + 255) / 256, 256, 0, stream>>>(Wqkv, wqkv16, n4);
    n4 = HID * HID / 4;
    cvt_f32_to_f16<<<(n4 + 255) / 256, 256, 0, stream>>>(Wout, wout16, n4);
  }
  // 2) qkv = x @ Wqkv^T + bqkv   (M=8192, N=3072, K=1024)
  gemm_f16<<<dim3(3 * HID / 64, NROWS / 64), 256, 0, stream>>>(
      x16, wqkv16, bqkv, qkv16, NROWS, 3 * HID, HID);
  // 3) QK norm + head split
  qk_norm<<<NROWS, 256, 0, stream>>>(qkv16, g_q, g_k, qh, kh, vh);
  // 4) attention
  attn_kernel<<<dim3(SEQ / 128, BATCH * NHEAD), 256, 0, stream>>>(qh, kh, vh, attn16);
  // 5) proj = attn @ Wout^T + bout  (M=8192, N=1024, K=1024)
  gemm_f16<<<dim3(HID / 64, NROWS / 64), 256, 0, stream>>>(
      attn16, wout16, bout, proj16, NROWS, HID, HID);
  // 6) final RMS norm -> fp32 out
  out_norm<<<NROWS, 256, 0, stream>>>(proj16, g_out, (float*)d_out);
}
